// PointVQVAE_15384572854333
// MI455X (gfx1250) — compile-verified, matrix operand order checked by reading, not run on hardware
//
#include <hip/hip_runtime.h>
#include <hip/hip_bf16.h>

// ---------------------------------------------------------------------------
// PointVQVAE for MI455X (gfx1250): bf16 WMMA GEMM backbone + wave32 helpers.
// ---------------------------------------------------------------------------

#define BN_RSQ 0.99999500003749968f   // 1/sqrt(1 + 1e-5)

typedef __attribute__((ext_vector_type(16))) __bf16 bf16x16;
typedef __attribute__((ext_vector_type(8)))  float  f32x8;

union FragBF { bf16x16 v; uint4 q[2]; };

// Load one 16x32 bf16 fragment (A-matrix layout; B uses the same gather with
// lane = output column since W is stored [N,K] row-major = B^T).
__device__ __forceinline__ bf16x16 pvq_load_tile16x32(const __hip_bfloat16* base,
                                                      int ld, int kb, int lane) {
  const int r  = lane & 15;
  const int hi = (lane >> 4) & 1;            // lanes 16..31 take K 8..15 / 24..31
  const __hip_bfloat16* p = base + (size_t)r * ld + kb + hi * 8;
  FragBF f;
  f.q[0] = *reinterpret_cast<const uint4*>(p);       // K block +0 / +8
  f.q[1] = *reinterpret_cast<const uint4*>(p + 16);  // K block +16 / +24
  return f.v;
}

// ---------------------------------------------------------------------------
// Generic fused GEMM: out = epilogue(A[M,K](bf16) * W[N,K]^T(bf16))
// epilogue: +bias, optional BN (g*rsqrt(1+eps), be), optional ReLU,
//           optional residual add; store bf16 or f32.
// grid = (M/32, N/128), block = 128 (4 waves).
// Each wave computes a 32x32 tile: 2 A-frags x 2 B-frags -> 4 independent
// v_wmma_f32_16x16x32_bf16 per K step (2 b128 loads per WMMA).
// ---------------------------------------------------------------------------
__global__ void pvq_gemm(const __hip_bfloat16* __restrict__ A, int lda,
                         const __hip_bfloat16* __restrict__ W, int ldw,
                         const float* __restrict__ bias,
                         const float* __restrict__ bng,
                         const float* __restrict__ bnbe,
                         const float* __restrict__ resid, int ldr,
                         int K, int mValid, int relu,
                         float* __restrict__ outF,
                         __hip_bfloat16* __restrict__ outB, int ldc) {
  const int lane = threadIdx.x & 31;
  const int wave = threadIdx.x >> 5;
  const int row0 = blockIdx.x << 5;                  // 32 rows per block
  const int col0 = (blockIdx.y << 7) + (wave << 5);  // 32 cols per wave

  const __hip_bfloat16* Ab0 = A + (size_t)row0 * lda;
  const __hip_bfloat16* Ab1 = A + (size_t)(row0 + 16) * lda;
  const __hip_bfloat16* Wb0 = W + (size_t)col0 * ldw;
  const __hip_bfloat16* Wb1 = W + (size_t)(col0 + 16) * ldw;

  f32x8 acc00, acc01, acc10, acc11;
#pragma unroll
  for (int z = 0; z < 8; ++z) {
    acc00[z] = 0.0f; acc01[z] = 0.0f; acc10[z] = 0.0f; acc11[z] = 0.0f;
  }

  for (int kb = 0; kb < K; kb += 32) {
    bf16x16 a0 = pvq_load_tile16x32(Ab0, lda, kb, lane);
    bf16x16 a1 = pvq_load_tile16x32(Ab1, lda, kb, lane);
    bf16x16 b0 = pvq_load_tile16x32(Wb0, ldw, kb, lane);
    bf16x16 b1 = pvq_load_tile16x32(Wb1, ldw, kb, lane);
    if (kb + 32 < K) {  // WGP-scope prefetch of next K block (global_prefetch_b8)
      __builtin_prefetch((const void*)(Ab0 + (size_t)(lane & 15) * lda + kb + 32), 0, 3);
      __builtin_prefetch((const void*)(Ab1 + (size_t)(lane & 15) * lda + kb + 32), 0, 3);
      __builtin_prefetch((const void*)(Wb0 + (size_t)(lane & 15) * ldw + kb + 32), 0, 3);
      __builtin_prefetch((const void*)(Wb1 + (size_t)(lane & 15) * ldw + kb + 32), 0, 3);
    }
    acc00 = __builtin_amdgcn_wmma_f32_16x16x32_bf16(false, a0, false, b0, (short)0, acc00, false, false);
    acc01 = __builtin_amdgcn_wmma_f32_16x16x32_bf16(false, a0, false, b1, (short)0, acc01, false, false);
    acc10 = __builtin_amdgcn_wmma_f32_16x16x32_bf16(false, a1, false, b0, (short)0, acc10, false, false);
    acc11 = __builtin_amdgcn_wmma_f32_16x16x32_bf16(false, a1, false, b1, (short)0, acc11, false, false);
  }

  const int rsub = (((lane >> 4) & 1) << 3);   // +8 for lanes 16..31
#pragma unroll
  for (int mt = 0; mt < 2; ++mt) {
#pragma unroll
    for (int nt = 0; nt < 2; ++nt) {
      f32x8 acc = mt ? (nt ? acc11 : acc10) : (nt ? acc01 : acc00);
      const int n = col0 + nt * 16 + (lane & 15);
      const float bi = bias ? bias[n] : 0.0f;
      float sc = 1.0f, sh = 0.0f;
      const bool hasbn = (bng != nullptr);
      if (hasbn) { sc = bng[n] * BN_RSQ; sh = bnbe[n]; }
#pragma unroll
      for (int i = 0; i < 8; ++i) {
        const int row = row0 + mt * 16 + rsub + i;
        if (row >= mValid) continue;
        float v = acc[i] + bi;
        if (hasbn) v = v * sc + sh;
        if (relu)  v = fmaxf(v, 0.0f);
        if (resid) v += resid[(size_t)row * ldr + n];
        if (outF) outF[(size_t)row * ldc + n] = v;
        else      outB[(size_t)row * ldc + n] = __float2bfloat16(v);
      }
    }
  }
}

// ---------------------------------------------------------------------------
// fp32 -> bf16 packing (weights / activations)
// ---------------------------------------------------------------------------
__global__ void pvq_pack_bf16(const float* __restrict__ src,
                              __hip_bfloat16* __restrict__ dst, int n) {
  int i = blockIdx.x * blockDim.x + threadIdx.x;
  if (i < n) dst[i] = __float2bfloat16(src[i]);
}

__global__ void pvq_pack_pad(const float* __restrict__ src,
                             __hip_bfloat16* __restrict__ dst,
                             int rows, int ks, int kd) {
  int i = blockIdx.x * blockDim.x + threadIdx.x;
  if (i >= rows * kd) return;
  int r = i / kd, k = i - r * kd;
  dst[i] = __float2bfloat16(k < ks ? src[(size_t)r * ks + k] : 0.0f);
}

// ---------------------------------------------------------------------------
// Farthest point sampling: one block per batch, sequential over M picks.
// Matches jax scan: idx[m] = last (before update), first pick = 0, argmax
// ties -> first index.
// ---------------------------------------------------------------------------
__global__ void pvq_fps(const float* __restrict__ xyz, int N, int M,
                        int* __restrict__ idx_out, float* __restrict__ newxyz) {
  __shared__ float dist[2048];
  __shared__ float rv[256];
  __shared__ int   ri[256];
  __shared__ float lp[3];
  __shared__ int   lastS;
  const int b = blockIdx.x, t = threadIdx.x;
  const float* P = xyz + (size_t)b * N * 3;
  for (int i = t; i < N; i += blockDim.x) dist[i] = 3.0e38f;
  if (t == 0) lastS = 0;
  __syncthreads();
  for (int m = 0; m < M; ++m) {
    const int last = lastS;
    if (t == 0) {
      idx_out[b * M + m] = last;
      newxyz[((size_t)b * M + m) * 3 + 0] = P[last * 3 + 0];
      newxyz[((size_t)b * M + m) * 3 + 1] = P[last * 3 + 1];
      newxyz[((size_t)b * M + m) * 3 + 2] = P[last * 3 + 2];
      lp[0] = P[last * 3 + 0]; lp[1] = P[last * 3 + 1]; lp[2] = P[last * 3 + 2];
    }
    __syncthreads();
    const float lx = lp[0], ly = lp[1], lz = lp[2];
    float bv = -1.0f; int bi = 0;
    for (int i = t; i < N; i += blockDim.x) {
      const float dx = P[i * 3 + 0] - lx;
      const float dy = P[i * 3 + 1] - ly;
      const float dz = P[i * 3 + 2] - lz;
      const float dd = fminf(dist[i], dx * dx + dy * dy + dz * dz);
      dist[i] = dd;
      if (dd > bv) { bv = dd; bi = i; }
    }
    rv[t] = bv; ri[t] = bi;
    __syncthreads();
    for (int s = blockDim.x >> 1; s > 0; s >>= 1) {
      if (t < s) {
        if (rv[t + s] > rv[t] || (rv[t + s] == rv[t] && ri[t + s] < ri[t])) {
          rv[t] = rv[t + s]; ri[t] = ri[t + s];
        }
      }
      __syncthreads();
    }
    if (t == 0) lastS = ri[0];
    __syncthreads();
  }
}

// ---------------------------------------------------------------------------
// Ball query: one wave32 per center; ballot-based stream compaction picks the
// first K in-radius points in index order; pad with first hit (0 if none).
// ---------------------------------------------------------------------------
__global__ void pvq_ball_query(const float* __restrict__ xyz,   // [B,N,3]
                               const float* __restrict__ ctr,   // [B,M,3]
                               int N, int Mcent, int K, float r2,
                               int* __restrict__ nidx, int numCenters) {
  const int w    = blockIdx.x * (blockDim.x >> 5) + (threadIdx.x >> 5);
  const int lane = threadIdx.x & 31;
  if (w >= numCenters) return;
  const int b = w / Mcent;
  const float* P = xyz + (size_t)b * N * 3;
  const float cx = ctr[w * 3 + 0], cy = ctr[w * 3 + 1], cz = ctr[w * 3 + 2];
  int* out = nidx + (size_t)w * K;
  int cnt = 0, firstIdx = -1;
  const unsigned lt = (1u << lane) - 1u;
  for (int base = 0; base < N; base += 32) {
    const int i = base + lane;
    const float dx = P[i * 3 + 0] - cx;
    const float dy = P[i * 3 + 1] - cy;
    const float dz = P[i * 3 + 2] - cz;
    const bool hit = (dx * dx + dy * dy + dz * dz) < r2;
    const unsigned msk = (unsigned)__ballot(hit);
    if (hit) {
      const int pos = cnt + __popc(msk & lt);
      if (pos < K) out[pos] = i;
    }
    if (firstIdx < 0 && msk) firstIdx = base + __builtin_ctz(msk);
    cnt += __popc(msk);
    if (cnt >= K) break;
  }
  if (firstIdx < 0) firstIdx = 0;
  for (int j = cnt + lane; j < K; j += 32) out[j] = firstIdx;
}

// ---------------------------------------------------------------------------
// SA1 first conv (K=6 is too small for WMMA): fused gather + conv + BN + ReLU
// -> bf16 rows [chunkCenters*64, 512] feeding the big WMMA GEMM.
// ---------------------------------------------------------------------------
__global__ void pvq_sa1_conv1(const float* __restrict__ pts,   // [8,2048,3]
                              const float* __restrict__ xyz1,  // [8,512,3]
                              const int*   __restrict__ nidx1, // [4096,64]
                              const float* __restrict__ Wc, const float* __restrict__ bc,
                              const float* __restrict__ gc, const float* __restrict__ bec,
                              int centerBase,
                              __hip_bfloat16* __restrict__ out) {
  const int rowLocal = blockIdx.x;
  const int c  = centerBase + (rowLocal >> 6);
  const int kk = rowLocal & 63;
  const int b  = c >> 9;
  const int p  = nidx1[(size_t)c * 64 + kk];
  const float* pp = pts  + ((size_t)b * 2048 + p) * 3;
  const float* cc = xyz1 + (size_t)c * 3;
  const float in0 = pp[0] - cc[0], in1 = pp[1] - cc[1], in2 = pp[2] - cc[2];
  const float in3 = pp[0], in4 = pp[1], in5 = pp[2];
  for (int o = threadIdx.x; o < 512; o += blockDim.x) {
    const float* w = Wc + (size_t)o * 6;
    float v = w[0] * in0 + w[1] * in1 + w[2] * in2 +
              w[3] * in3 + w[4] * in4 + w[5] * in5 + bc[o];
    v = v * (gc[o] * BN_RSQ) + bec[o];
    out[(size_t)rowLocal * 512 + o] = __float2bfloat16(fmaxf(v, 0.0f));
  }
}

// SA2 grouped-feature builder: rows [chunk*512, 544] bf16 (515 data + 29 zero pad)
__global__ void pvq_build_g2(const float* __restrict__ xyz1,  // [8,512,3]
                             const float* __restrict__ xyz2,  // [8,16,3]
                             const float* __restrict__ feat1, // [4096,512]
                             const int* __restrict__ nidx2,   // [128,512]
                             int centerBase,
                             __hip_bfloat16* __restrict__ out) {
  const int rowLocal = blockIdx.x;
  const int c = centerBase + (rowLocal >> 9);
  const int j = rowLocal & 511;
  const int b = c >> 4;
  const int p = nidx2[(size_t)c * 512 + j];
  const float* pxyz = xyz1 + ((size_t)b * 512 + p) * 3;
  const float* cc   = xyz2 + (size_t)c * 3;
  const float* f    = feat1 + ((size_t)b * 512 + p) * 512;
  __hip_bfloat16* o = out + (size_t)rowLocal * 544;
  for (int d = threadIdx.x; d < 544; d += blockDim.x) {
    float v;
    if (d < 3)        v = pxyz[d] - cc[d];
    else if (d < 515) v = f[d - 3];
    else              v = 0.0f;
    o[d] = __float2bfloat16(v);
  }
}

// Max over G consecutive bf16 rows -> f32 row.
__global__ void pvq_rowmax(const __hip_bfloat16* __restrict__ in, int G, int C,
                           float* __restrict__ out) {
  const int g = blockIdx.x;
  for (int col = threadIdx.x; col < C; col += blockDim.x) {
    float m = -3.0e38f;
    const __hip_bfloat16* p = in + (size_t)g * G * C + col;
    for (int r = 0; r < G; ++r) m = fmaxf(m, __bfloat162float(p[(size_t)r * C]));
    out[(size_t)g * C + col] = m;
  }
}

// Vector quantization: one block per token; argmin over 512 codes (ties->first).
__global__ void pvq_vq(const float* __restrict__ feat2, // [128,512]
                       const float* __restrict__ cbk,   // [512,512]
                       float* __restrict__ x, __hip_bfloat16* __restrict__ xq) {
  __shared__ float fsh[512];
  __shared__ float rv[256];
  __shared__ int   ri[256];
  const int t = blockIdx.x, tid = threadIdx.x;
  const float* fr = feat2 + (size_t)t * 512;
  for (int d = tid; d < 512; d += 256) fsh[d] = fr[d];
  __syncthreads();
  float bv = 3.0e38f; int bi = 0;
  for (int code = tid; code < 512; code += 256) {
    const float* c = cbk + (size_t)code * 512;
    float s = 0.0f, n2 = 0.0f;
    for (int d = 0; d < 512; ++d) { const float cv = c[d]; s += fsh[d] * cv; n2 += cv * cv; }
    const float dist = n2 - 2.0f * s;   // ||f||^2 constant per token
    if (dist < bv || (dist == bv && code < bi)) { bv = dist; bi = code; }
  }
  rv[tid] = bv; ri[tid] = bi;
  __syncthreads();
  for (int s2 = 128; s2 > 0; s2 >>= 1) {
    if (tid < s2) {
      if (rv[tid + s2] < rv[tid] || (rv[tid + s2] == rv[tid] && ri[tid + s2] < ri[tid])) {
        rv[tid] = rv[tid + s2]; ri[tid] = ri[tid + s2];
      }
    }
    __syncthreads();
  }
  const int best = ri[0];
  const float* cbest = cbk + (size_t)best * 512;
  for (int d = tid; d < 512; d += 256) {
    const float v = cbest[d];                 // straight-through fwd == quant
    x[(size_t)t * 512 + d]  = v;
    xq[(size_t)t * 512 + d] = __float2bfloat16(v);
  }
}

// Max over 16 tokens per batch + ReLU; rows 8..31 zero-padded for 32-row GEMM.
__global__ void pvq_pool(const float* __restrict__ x, // [128,512]
                         __hip_bfloat16* __restrict__ pooledB) {
  const int r = blockIdx.x;  // 0..31
  for (int col = threadIdx.x; col < 512; col += blockDim.x) {
    float v = 0.0f;
    if (r < 8) {
      float m = -3.0e38f;
      for (int tkn = 0; tkn < 16; ++tkn)
        m = fmaxf(m, x[((size_t)r * 16 + tkn) * 512 + col]);
      v = fmaxf(m, 0.0f);
    }
    pooledB[(size_t)r * 512 + col] = __float2bfloat16(v);
  }
}

// ---------------------------------------------------------------------------
// Host orchestration
// ---------------------------------------------------------------------------
extern "C" void kernel_launch(void* const* d_in, const int* in_sizes, int n_in,
                              void* d_out, int out_size, void* d_ws, size_t ws_size,
                              hipStream_t stream) {
  (void)out_size; (void)ws_size;
  // Leaf order: insertion order (pts first). Fallback: jax.tree_leaves
  // alphabetical order (codebook first) detected via in_sizes[0].
  const bool alpha = (n_in > 0) && (in_sizes[0] != 8 * 2048 * 3);
  auto IN = [&](int ins, int alp) -> const float* {
    return (const float*)d_in[alpha ? alp : ins];
  };
  const float* pts   = IN(0, 53);
  const float* s1aW  = IN(1, 37);  const float* s1ab = IN(2, 38);
  const float* s1ag  = IN(3, 40);  const float* s1abe = IN(4, 39);
  const float* s1bW  = IN(5, 41);  const float* s1bb = IN(6, 42);
  const float* s1bg  = IN(7, 44);  const float* s1bbe = IN(8, 43);
  const float* s2aW  = IN(9, 45);  const float* s2ab = IN(10, 46);
  const float* s2ag  = IN(11, 48); const float* s2abe = IN(12, 47);
  const float* s2bW  = IN(13, 49); const float* s2bb = IN(14, 50);
  const float* s2bg  = IN(15, 52); const float* s2bbe = IN(16, 51);
  const float* cbk   = IN(17, 0);
  const float *faW[3], *fab[3], *fag[3], *fabe[3];
  const float *fbW[3], *fbb[3], *fbg[3], *fbbe[3];
  const float *fcW[3], *fcb[3];
  for (int f = 0; f < 3; ++f) {
    const int bi = 18 + 10 * f, ba = 7 + 10 * f;
    faW[f] = IN(bi + 0, ba + 0); fab[f] = IN(bi + 1, ba + 1);
    fag[f] = IN(bi + 2, ba + 3); fabe[f] = IN(bi + 3, ba + 2);
    fbW[f] = IN(bi + 4, ba + 4); fbb[f] = IN(bi + 5, ba + 5);
    fbg[f] = IN(bi + 6, ba + 7); fbbe[f] = IN(bi + 7, ba + 6);
    fcW[f] = IN(bi + 8, ba + 8); fcb[f] = IN(bi + 9, ba + 9);
  }
  const float* e0W = IN(48, 1); const float* e0b = IN(49, 2);
  const float* e0g = IN(50, 4); const float* e0be = IN(51, 3);
  const float* e1W = IN(52, 5); const float* e1b = IN(53, 6);

  // Workspace carve (256B aligned), ~125 MiB total.
  char* wsp = (char*)d_ws; size_t off = 0;
  auto carve = [&](size_t bytes) -> void* {
    void* p = wsp + off; off = (off + bytes + 255) & ~(size_t)255; return p;
  };
  __hip_bfloat16* w_s1b = (__hip_bfloat16*)carve((size_t)512 * 512 * 2);
  __hip_bfloat16* w_s2a = (__hip_bfloat16*)carve((size_t)512 * 544 * 2);
  __hip_bfloat16* w_s2b = (__hip_bfloat16*)carve((size_t)512 * 512 * 2);
  __hip_bfloat16 *w_fa[3], *w_fb[3], *w_fc[3];
  for (int f = 0; f < 3; ++f) {
    w_fa[f] = (__hip_bfloat16*)carve((size_t)2048 * 512 * 2);
    w_fb[f] = (__hip_bfloat16*)carve((size_t)2048 * 2048 * 2);
    w_fc[f] = (__hip_bfloat16*)carve((size_t)512 * 2048 * 2);
  }
  __hip_bfloat16* w_e0 = (__hip_bfloat16*)carve((size_t)1024 * 512 * 2);
  __hip_bfloat16* w_e1 = (__hip_bfloat16*)carve((size_t)6144 * 1024 * 2);
  int*   idx1  = (int*)carve((size_t)8 * 512 * 4);
  float* xyz1  = (float*)carve((size_t)8 * 512 * 3 * 4);
  int*   nidx1 = (int*)carve((size_t)8 * 512 * 64 * 4);
  int*   idx2  = (int*)carve((size_t)8 * 16 * 4);
  float* xyz2  = (float*)carve((size_t)8 * 16 * 3 * 4);
  int*   nidx2 = (int*)carve((size_t)8 * 16 * 512 * 4);
  float* feat1 = (float*)carve((size_t)4096 * 512 * 4);
  float* feat2 = (float*)carve((size_t)128 * 512 * 4);
  __hip_bfloat16* bufA = (__hip_bfloat16*)carve((size_t)32768 * 512 * 2);
  __hip_bfloat16* bufB = (__hip_bfloat16*)carve((size_t)32768 * 512 * 2);
  float* x0 = (float*)carve((size_t)128 * 512 * 4);
  float* x1 = (float*)carve((size_t)128 * 512 * 4);
  __hip_bfloat16* xq      = (__hip_bfloat16*)carve((size_t)128 * 512 * 2);
  __hip_bfloat16* pooledB = (__hip_bfloat16*)carve((size_t)32 * 512 * 2);
  __hip_bfloat16* hend    = (__hip_bfloat16*)carve((size_t)32 * 1024 * 2);

  auto packN = [&](const float* s, __hip_bfloat16* d, size_t n) {
    pvq_pack_bf16<<<dim3((unsigned)((n + 255) / 256)), 256, 0, stream>>>(s, d, (int)n);
  };
  // Pack all GEMM weights to bf16 (stays resident in 192MB L2 thereafter).
  packN(s1bW, w_s1b, (size_t)512 * 512);
  pvq_pack_pad<<<dim3((512 * 544 + 255) / 256), 256, 0, stream>>>(s2aW, w_s2a, 512, 515, 544);
  packN(s2bW, w_s2b, (size_t)512 * 512);
  for (int f = 0; f < 3; ++f) {
    packN(faW[f], w_fa[f], (size_t)2048 * 512);
    packN(fbW[f], w_fb[f], (size_t)2048 * 2048);
    packN(fcW[f], w_fc[f], (size_t)512 * 2048);
  }
  packN(e0W, w_e0, (size_t)1024 * 512);
  packN(e1W, w_e1, (size_t)6144 * 1024);

  // ---- SA1: FPS(2048->512), ball query r=0.1 k=64, conv 6->512, WMMA 512->512, max64
  pvq_fps<<<8, 256, 0, stream>>>(pts, 2048, 512, idx1, xyz1);
  pvq_ball_query<<<512, 256, 0, stream>>>(pts, xyz1, 2048, 512, 64, 0.01f, nidx1, 4096);
  for (int ck = 0; ck < 8; ++ck) {
    const int cbase = ck * 512;
    pvq_sa1_conv1<<<32768, 128, 0, stream>>>(pts, xyz1, nidx1, s1aW, s1ab, s1ag, s1abe,
                                             cbase, bufA);
    pvq_gemm<<<dim3(1024, 4), 128, 0, stream>>>(bufA, 512, w_s1b, 512,
                                                s1bb, s1bg, s1bbe, nullptr, 0,
                                                512, 32768, 1, nullptr, bufB, 512);
    pvq_rowmax<<<512, 256, 0, stream>>>(bufB, 64, 512, feat1 + (size_t)cbase * 512);
  }

  // ---- SA2: FPS(512->16), ball query r=0.4 k=512, WMMA 515(pad544)->512 ->512, max512
  pvq_fps<<<8, 256, 0, stream>>>(xyz1, 512, 16, idx2, xyz2);
  pvq_ball_query<<<16, 256, 0, stream>>>(xyz1, xyz2, 512, 16, 512, 0.16f, nidx2, 128);
  for (int ck = 0; ck < 8; ++ck) {
    const int cbase = ck * 16;
    pvq_build_g2<<<8192, 128, 0, stream>>>(xyz1, xyz2, feat1, nidx2, cbase, bufA);
    pvq_gemm<<<dim3(256, 4), 128, 0, stream>>>(bufA, 544, w_s2a, 544,
                                               s2ab, s2ag, s2abe, nullptr, 0,
                                               544, 8192, 1, nullptr, bufB, 512);
    pvq_gemm<<<dim3(256, 4), 128, 0, stream>>>(bufB, 512, w_s2b, 512,
                                               s2bb, s2bg, s2bbe, nullptr, 0,
                                               512, 8192, 1, nullptr, bufA, 512);
    pvq_rowmax<<<16, 256, 0, stream>>>(bufA, 512, 512, feat2 + (size_t)cbase * 512);
  }

  // ---- Vector quantization
  pvq_vq<<<128, 256, 0, stream>>>(feat2, cbk, x0, xq);

  // ---- Residual folds: x = x + fold(x)
  float* xc = x0; float* xn = x1;
  for (int f = 0; f < 3; ++f) {
    pvq_gemm<<<dim3(4, 16), 128, 0, stream>>>(xq, 512, w_fa[f], 512,
                                              fab[f], fag[f], fabe[f], nullptr, 0,
                                              512, 128, 1, nullptr, bufA, 2048);
    pvq_gemm<<<dim3(4, 16), 128, 0, stream>>>(bufA, 2048, w_fb[f], 2048,
                                              fbb[f], fbg[f], fbbe[f], nullptr, 0,
                                              2048, 128, 1, nullptr, bufB, 2048);
    pvq_gemm<<<dim3(4, 4), 128, 0, stream>>>(bufB, 2048, w_fc[f], 2048,
                                             fcb[f], nullptr, nullptr, xc, 512,
                                             2048, 128, 0, xn, nullptr, 512);
    packN(xn, xq, (size_t)128 * 512);
    float* tmp = xc; xc = xn; xn = tmp;
  }

  // ---- Global max over 16 tokens + ReLU, then head (rows padded 8->32)
  pvq_pool<<<32, 256, 0, stream>>>(xc, pooledB);
  pvq_gemm<<<dim3(1, 8), 128, 0, stream>>>(pooledB, 512, w_e0, 512,
                                           e0b, e0g, e0be, nullptr, 0,
                                           512, 32, 1, nullptr, hend, 1024);
  pvq_gemm<<<dim3(1, 48), 128, 0, stream>>>(hend, 1024, w_e1, 1024,
                                            e1b, nullptr, nullptr, nullptr, 0,
                                            1024, 8, 0, (float*)d_out, nullptr, 6144);
}
